// LSTM_45732811767992
// MI455X (gfx1250) — matrix operand order repaired and checked by reading, not compiled, not run on hardware
//
#include <hip/hip_runtime.h>
#include <hip/hip_bf16.h>

typedef __attribute__((ext_vector_type(16))) __bf16 v16bf;
typedef __attribute__((ext_vector_type(8)))  __bf16 v8bf;
typedef __attribute__((ext_vector_type(8)))  float  v8f;

#define T_STEPS 128
#define IN_DIM  64
#define H_DIM   256
#define G_DIM   1024   // 4*H
#define TILE_B  32
#define NTHREADS 512   // 16 waves
#define GSTR    1028   // padded f32 row stride for gates LDS (bank-conflict free scatter)
#define HSTR    272    // padded bf16 row stride for h LDS
#define XSTR    80     // padded bf16 row stride for x LDS

// workspace layout (bytes)
#define WS_WHH  0
#define WS_WIH  (G_DIM * H_DIM * 2)            // 524288
#define WS_BC   (WS_WIH + G_DIM * IN_DIM * 2)  // 655360

// ---------- prep: f32 -> bf16 weights, combine biases ----------
__global__ void prep_kernel(const float* __restrict__ Wih,
                            const float* __restrict__ Whh,
                            const float* __restrict__ bih,
                            const float* __restrict__ bhh,
                            __bf16* __restrict__ wih_bf,
                            __bf16* __restrict__ whh_bf,
                            float* __restrict__ bc) {
  int idx = blockIdx.x * blockDim.x + threadIdx.x;
  const int nWih = G_DIM * IN_DIM;
  const int nWhh = G_DIM * H_DIM;
  if (idx < nWih) {
    wih_bf[idx] = (__bf16)Wih[idx];
  } else if (idx < nWih + nWhh) {
    int k = idx - nWih;
    whh_bf[k] = (__bf16)Whh[k];
  } else if (idx < nWih + nWhh + G_DIM) {
    int k = idx - nWih - nWhh;
    bc[k] = bih[k] + bhh[k];
  }
}

// A-tile (16x32 bf16) loader, contiguous k order (same as the B-tile loads):
// lanes 0-15: M=0..15, K = k0..k0+15; lanes 16-31: K = k0+16..k0+31
__device__ __forceinline__ v16bf load_a_tile(const __bf16* base, int stride,
                                             int r0, int k0, int lane) {
  int m = lane & 15, half = lane >> 4;
  const __bf16* p = base + (r0 + m) * stride;
  union { v16bf v; v8bf h[2]; } a;
  a.h[0] = *(const v8bf*)(p + k0 + 16 * half);
  a.h[1] = *(const v8bf*)(p + k0 + 16 * half + 8);
  return a.v;
}

// ---------- fused LSTM: one workgroup (16 waves) per 32 batch rows ----------
__global__ __launch_bounds__(NTHREADS)
void lstm_fused_kernel(const float* __restrict__ x,
                       const __bf16* __restrict__ wih,
                       const __bf16* __restrict__ whh,
                       const float* __restrict__ bc,
                       const float* __restrict__ fcW,
                       const float* __restrict__ fcb,
                       float* __restrict__ out, int NC) {
  extern __shared__ char smem[];
  float*  gates = (float*)smem;                                            // [32][GSTR] f32
  __bf16* hbuf  = (__bf16*)(smem + TILE_B * GSTR * 4);                     // [32][HSTR] bf16
  __bf16* xbuf  = (__bf16*)(smem + TILE_B * GSTR * 4 + TILE_B * HSTR * 2); // [32][XSTR]

  const int tid  = threadIdx.x;
  const int lane = tid & 31;
  const int wid  = tid >> 5;          // 16 waves
  const int b0   = blockIdx.x * TILE_B;
  const int colbase = wid * 64;       // this wave's 64 gate columns (4 col-tiles)

  // elementwise ownership: column j, rows [rhalf*16, rhalf*16+16)
  const int j     = tid & 255;
  const int rhalf = tid >> 8;         // 0 or 1

  // t-invariant per-lane 32-bit element offsets for B-tile loads
  // (B[k][n] = W[n0+n, k]; lane: col n = lane%16, 16 contiguous K at k0+16*(lane/16))
  const int nsel = lane & 15, khalf = lane >> 4;
  const int off_whh_base = (colbase + nsel) * H_DIM + 16 * khalf;
  const int off_wih_base = (colbase + nsel) * IN_DIM + 16 * khalf;

  // init h = 0 in LDS, c = 0 in registers
  for (int i = tid; i < TILE_B * HSTR; i += NTHREADS) hbuf[i] = (__bf16)0.0f;
  float c_reg[16];
#pragma unroll
  for (int u = 0; u < 16; ++u) c_reg[u] = 0.0f;

  // per-thread biases (t-invariant)
  const float bi = bc[j], bf_ = bc[256 + j], bg = bc[512 + j], bo = bc[768 + j];

  __syncthreads();

  for (int t = 0; t < T_STEPS; ++t) {
    // Opaque per-lane voffsets, "redefined" every iteration: keeps the weight
    // B-tile loads inside the loop (no LICM -> no spill) while the base
    // pointers stay clean SGPR pairs -> global_load_b128 saddr+voffset+imm,
    // tracked by LOADcnt only, served from L2 shared by all workgroups.
    int owhh = off_whh_base;
    int owih = off_wih_base;
    asm volatile("" : "+v"(owhh), "+v"(owih));

    // ---- stage x[:, t, :] tile -> bf16 LDS (coalesced, 32-bit offsets) ----
#pragma unroll
    for (int r = 0; r < 4; ++r) {
      int e = tid + NTHREADS * r;
      int row = e >> 6, col = e & 63;
      int xoff = ((b0 + row) * T_STEPS + t) * IN_DIM + col;
      xbuf[row * XSTR + col] = (__bf16)x[xoff];
    }
    __syncthreads();

    // ---- gates = x_t @ Wih^T + h @ Whh^T ----
    v8f acc[2][4];
#pragma unroll
    for (int rt = 0; rt < 2; ++rt)
#pragma unroll
      for (int ct = 0; ct < 4; ++ct)
#pragma unroll
        for (int q = 0; q < 8; ++q) acc[rt][ct][q] = 0.0f;

#pragma unroll
    for (int kt = 0; kt < 2; ++kt) {            // K = 64 input GEMM
      v16bf a0 = load_a_tile(xbuf, XSTR, 0,  kt * 32, lane);
      v16bf a1 = load_a_tile(xbuf, XSTR, 16, kt * 32, lane);
#pragma unroll
      for (int ct = 0; ct < 4; ++ct) {
        // constant delta folds into the 24-bit instruction offset
        v16bf b = *(const v16bf*)(wih + owih + ct * (16 * IN_DIM) + kt * 32);
        acc[0][ct] = __builtin_amdgcn_wmma_f32_16x16x32_bf16(
            false, a0, false, b, (short)0, acc[0][ct], false, false);
        acc[1][ct] = __builtin_amdgcn_wmma_f32_16x16x32_bf16(
            false, a1, false, b, (short)0, acc[1][ct], false, false);
      }
    }
#pragma unroll
    for (int kt = 0; kt < 8; ++kt) {            // K = 256 recurrent GEMM (B streamed from L2)
      v16bf a0 = load_a_tile(hbuf, HSTR, 0,  kt * 32, lane);
      v16bf a1 = load_a_tile(hbuf, HSTR, 16, kt * 32, lane);
#pragma unroll
      for (int ct = 0; ct < 4; ++ct) {
        v16bf b = *(const v16bf*)(whh + owhh + ct * (16 * H_DIM) + kt * 32);
        acc[0][ct] = __builtin_amdgcn_wmma_f32_16x16x32_bf16(
            false, a0, false, b, (short)0, acc[0][ct], false, false);
        acc[1][ct] = __builtin_amdgcn_wmma_f32_16x16x32_bf16(
            false, a1, false, b, (short)0, acc[1][ct], false, false);
      }
    }

    // ---- scatter accumulators to gates LDS (C layout: row = r + 8*(L/16)) ----
#pragma unroll
    for (int rt = 0; rt < 2; ++rt)
#pragma unroll
      for (int ct = 0; ct < 4; ++ct) {
        int cc = colbase + ct * 16 + (lane & 15);
        int rbase = rt * 16 + 8 * (lane >> 4);
#pragma unroll
        for (int r = 0; r < 8; ++r)
          gates[(rbase + r) * GSTR + cc] = acc[rt][ct][r];
      }
    __syncthreads();

    // ---- elementwise cell update: thread = (column j, 16 rows) ----
#pragma unroll
    for (int uu = 0; uu < 16; ++uu) {
      int u = rhalf * 16 + uu;
      float gi = gates[u * GSTR + j]        + bi;
      float gf = gates[u * GSTR + 256 + j]  + bf_;
      float gg = gates[u * GSTR + 512 + j]  + bg;
      float go = gates[u * GSTR + 768 + j]  + bo;
      float i_ = 1.0f / (1.0f + __expf(-gi));
      float f_ = 1.0f / (1.0f + __expf(-gf));
      float g_ = tanhf(gg);
      float o_ = 1.0f / (1.0f + __expf(-go));
      float c  = f_ * c_reg[uu] + i_ * g_;
      c_reg[uu] = c;
      hbuf[u * HSTR + j] = (__bf16)(o_ * tanhf(c));
    }
    __syncthreads();
  }

  // ---- fused FC: out[b, nc] = h_last[b] . fcW[nc] + fcb[nc]  (NC==1) ----
  if (tid < TILE_B) {
    int b = tid;
    for (int nc = 0; nc < NC; ++nc) {
      float s = fcb[nc];
      for (int jj = 0; jj < H_DIM; ++jj)
        s += (float)hbuf[b * HSTR + jj] * fcW[nc * H_DIM + jj];
      out[(b0 + b) * NC + nc] = s;
    }
  }
}

extern "C" void kernel_launch(void* const* d_in, const int* in_sizes, int n_in,
                              void* d_out, int out_size, void* d_ws, size_t ws_size,
                              hipStream_t stream) {
  const float* x   = (const float*)d_in[0];
  const float* Wih = (const float*)d_in[1];
  const float* Whh = (const float*)d_in[2];
  const float* bih = (const float*)d_in[3];
  const float* bhh = (const float*)d_in[4];
  const float* fcW = (const float*)d_in[5];
  const float* fcb = (const float*)d_in[6];
  float* out = (float*)d_out;

  int B  = in_sizes[0] / (T_STEPS * IN_DIM);   // 2048
  int NC = in_sizes[5] / H_DIM;                // 1

  __bf16* whh_bf = (__bf16*)((char*)d_ws + WS_WHH);
  __bf16* wih_bf = (__bf16*)((char*)d_ws + WS_WIH);
  float*  bc     = (float*)((char*)d_ws + WS_BC);

  int total = G_DIM * IN_DIM + G_DIM * H_DIM + G_DIM;
  prep_kernel<<<(total + 255) / 256, 256, 0, stream>>>(Wih, Whh, bih, bhh,
                                                       wih_bf, whh_bf, bc);

  size_t smem = (size_t)TILE_B * GSTR * 4 + (size_t)TILE_B * HSTR * 2 +
                (size_t)TILE_B * XSTR * 2;     // ~154 KB < 320 KB/WGP
  lstm_fused_kernel<<<B / TILE_B, NTHREADS, smem, stream>>>(x, wih_bf, whh_bf, bc,
                                                            fcW, fcb, out, NC);
}
